// LatentTokenDecoder_4372276707820
// MI455X (gfx1250) — matrix operand order repaired and checked by reading, not compiled, not run on hardware
//
#include <hip/hip_runtime.h>

typedef __attribute__((ext_vector_type(16))) _Float16 v16h;
typedef __attribute__((ext_vector_type(8)))  _Float16 v8h;
typedef __attribute__((ext_vector_type(8)))  float    v8f;
typedef __attribute__((ext_vector_type(4)))  unsigned int u32x4;
typedef __attribute__((ext_vector_type(8)))  int      i32x8;
typedef __attribute__((ext_vector_type(4)))  int      i32x4;

#define NLAYERS 6
#define BATCH   32

// ---------------------------------------------------------------------------
// style[b][c] = sum_j x[b][j] * fc_w[c][j] + fc_b[c]      (32x512, K=256)
// ---------------------------------------------------------------------------
__global__ void style_kernel(const float* __restrict__ x,
                             const float* __restrict__ fcw,
                             const float* __restrict__ fcb,
                             float* __restrict__ style) {
    int i = blockIdx.x * blockDim.x + threadIdx.x;
    if (i >= BATCH * 512) return;
    int b = i >> 9, c = i & 511;
    const float* xr = x + b * 256;
    const float* wr = fcw + c * 256;
    float acc = fcb[c];
    for (int j = 0; j < 256; ++j) acc += xr[j] * wr[j];
    style[i] = acc;
}

// ---------------------------------------------------------------------------
// s[b][c] = sum_j style[b][j] * mod_w[c][j] + mod_b[c]    (32xCin, K=512)
// ---------------------------------------------------------------------------
__global__ void mod_kernel(const float* __restrict__ style,
                           const float* __restrict__ mw,
                           const float* __restrict__ mb,
                           float* __restrict__ s, int Cin) {
    int i = blockIdx.x * blockDim.x + threadIdx.x;
    if (i >= BATCH * Cin) return;
    int b = i / Cin, c = i - b * Cin;
    const float* sr = style + b * 512;
    const float* wr = mw + c * 512;
    float acc = mb[c];
    for (int j = 0; j < 512; ++j) acc += sr[j] * wr[j];
    s[i] = acc;
}

// ---------------------------------------------------------------------------
// Pack conv weights (Cout, Cin, 3, 3) f32 -> row-major f16 [Cout][Kpad],
// K = Cin*9 zero-padded to Kpad (mult of 32): branch-free A, TDM-friendly rows.
// ---------------------------------------------------------------------------
__global__ void pack_w_kernel(const float* __restrict__ w,
                              _Float16* __restrict__ wf,
                              int Cout, int Kreal, int Kpad) {
    int i = blockIdx.x * blockDim.x + threadIdx.x;
    if (i >= Cout * Kpad) return;
    int m = i / Kpad, k = i - m * Kpad;
    wf[i] = (k < Kreal) ? (_Float16)w[m * Kreal + k] : (_Float16)0.0f;
}

__global__ void zero_kernel(float* __restrict__ p, int n) {
    int i = blockIdx.x * blockDim.x + threadIdx.x;
    if (i < n) p[i] = 0.0f;
}

// ---------------------------------------------------------------------------
// Implicit-GEMM 3x3 conv, fully specialized per layer.
//  M = Cout (16/wave), N = B*H*W pixels (16/tile, tiles never cross an image
//  since H*W % 16 == 0), K = Cin*9.
//  B side: modulated f16 activation strip (halo, zero padded) staged in LDS.
//  A side: weight tiles (16 x 32 f16, row stride Kpad) streamed into per-wave
//          LDS slots by the Tensor Data Mover, double buffered, synchronized
//          with s_wait_tensorcnt. A-fragments read as 2x ds_load_b128.
//  Epilogue: +bias, store conv, shfl-reduced sum(x^2) -> demod stats.
// ---------------------------------------------------------------------------
template<int CIN, int COUT, int H, int W, int MT>
__global__ __launch_bounds__(32 * MT)
void conv_wmma_kernel(const float* __restrict__ act, int actBS,
                      const float* __restrict__ s,
                      const _Float16* __restrict__ Wf16,
                      const float* __restrict__ convB,
                      float* __restrict__ convBuf,
                      float* __restrict__ sumsq) {
    constexpr int SW    = (W < 16) ? W : 16;
    constexpr int R     = 16 / SW;
    constexpr int KREAL = CIN * 9;
    constexpr int KPAD  = ((KREAL + 31) / 32) * 32;
    constexpr int SROW  = SW + 2, NR = R + 2;
    constexpr int BTOT  = CIN * NR * SROW;
    constexpr int BPAD  = (BTOT + 31) & ~31;          // 64B-align A region
    constexpr int HW    = H * W;

    extern __shared__ _Float16 smem[];
    _Float16* bstrip = smem;
    _Float16* astage = smem + BPAD;                   // MT waves * 2 slots * 512 h

    const int tid = threadIdx.x;
    const int p0  = blockIdx.x * 16;
    const int b   = p0 / HW;
    const int rem = p0 - b * HW;
    const int y0  = rem / W;
    const int x0  = rem - y0 * W;

    const int lane = tid & 31;
    const int wid  = tid >> 5;
    const int mt0  = (blockIdx.y * MT + wid) * 16;

    // ---- kick off first TDM weight tile before B staging (overlap) ----
    const unsigned ldsAbase = (unsigned)(uintptr_t)(astage + wid * 1024);
    auto tdm_load_A = [&](int k0, int slot) {
        unsigned long long ga =
            (unsigned long long)(uintptr_t)(Wf16 + (size_t)mt0 * KPAD + k0);
        u32x4 g0;
        g0[0] = 1u;                                   // count=1 (valid user D#)
        g0[1] = ldsAbase + (unsigned)slot * 1024u;    // lds_addr (bytes)
        g0[2] = (unsigned)(ga & 0xffffffffu);         // global_addr[31:0]
        g0[3] = (unsigned)((ga >> 32) & 0x1ffffffu)   // global_addr[56:32]
              | (2u << 30);                           // type = 2 ("image")
        i32x8 g1;
        g1[0] = (1 << 16);                            // wg_mask=0, data_size=2B
        g1[1] = (int)(((unsigned)KPAD & 0xffffu) << 16);       // tensor_dim0 lo
        g1[2] = (int)(((unsigned)KPAD >> 16) |
                      (((unsigned)COUT & 0xffffu) << 16));     // td0 hi | td1 lo
        g1[3] = (int)(((unsigned)COUT >> 16) | (32u << 16));   // td1 hi | tile_dim0=32
        g1[4] = 16;                                   // tile_dim1=16, tile_dim2=0
        g1[5] = KPAD;                                 // tensor_dim0_stride[31:0]
        g1[6] = 0;                                    // stride0 hi | stride1 lo
        g1[7] = 0;                                    // stride1 hi
        i32x4 z4 = {0, 0, 0, 0};
        i32x8 z8 = {0, 0, 0, 0, 0, 0, 0, 0};
        __builtin_amdgcn_tensor_load_to_lds(g0, g1, z4, z4, z8, 0);
    };
    tdm_load_A(0, 0);

    // ---- stage modulated f16 activation strip (incl. halo, zero padded) ----
    for (int e = tid; e < BTOT; e += 32 * MT) {
        int cin = e / (NR * SROW);
        int r2  = e - cin * NR * SROW;
        int dy  = r2 / SROW;
        int dx  = r2 - dy * SROW;
        int iy  = y0 - 1 + dy, ix = x0 - 1 + dx;
        float v = 0.0f;
        if (iy >= 0 && iy < H && ix >= 0 && ix < W)
            v = act[(size_t)b * actBS + ((size_t)cin * H + iy) * W + ix]
                * s[b * CIN + cin];
        bstrip[e] = (_Float16)v;
    }
    __syncthreads();

    const int n   = lane & 15;
    const int py  = n / SW, px = n - py * SW;
    const int kbA = (lane < 16) ? 0 : 8;     // A 16x32 f16 lane layout
    const int kbB = (lane < 16) ? 0 : 8;     // B 32x16: same k order as A (8-runs)

    v8f acc = {};
    int slot = 0;
    for (int k0 = 0; k0 < KREAL; k0 += 32) {
        if (k0 + 32 < KREAL) {
            tdm_load_A(k0 + 32, slot ^ 1);
            __builtin_amdgcn_s_wait_tensorcnt((short)1);
        } else {
            __builtin_amdgcn_s_wait_tensorcnt((short)0);
        }

        // A fragment: two contiguous 8-half runs per lane from the LDS tile
        const _Float16* arow = astage + wid * 1024 + slot * 512 + (lane & 15) * 32;
        v8h lo = *(const v8h*)(arow + kbA);
        v8h hi = *(const v8h*)(arow + 16 + kbA);
        v16h afrag;
#pragma unroll
        for (int e = 0; e < 8; ++e) { afrag[e] = lo[e]; afrag[8 + e] = hi[e]; }

        // B fragment: im2col gather from the staged strip
        v16h bfrag;
#pragma unroll
        for (int v = 0; v < 8; ++v) {
#pragma unroll
            for (int h = 0; h < 2; ++h) {
                int k = k0 + kbB + 2 * v + h + ((v >= 4) ? 8 : 0);
                _Float16 val = (_Float16)0.0f;
                if (KREAL % 32 == 0 || k < KREAL) {
                    int cin = k / 9;
                    int r   = k - cin * 9;
                    int ry  = r / 3, rx = r - ry * 3;
                    val = bstrip[(cin * NR + (py + ry)) * SROW + (px + rx)];
                }
                bfrag[2 * v + h] = val;
            }
        }

        acc = __builtin_amdgcn_wmma_f32_16x16x32_f16(
            false, afrag, false, bfrag, (short)0, acc, false, false);
        slot ^= 1;
    }

    // ---- epilogue: bias, store, demod sum-of-squares ----
    const int oy = y0 + py, ox = x0 + px;
#pragma unroll
    for (int v = 0; v < 8; ++v) {
        int m    = (lane < 16) ? v : (v + 8);
        int cout = mt0 + m;
        float val = acc[v] + convB[cout];
        convBuf[(((size_t)b * COUT + cout) * H + oy) * W + ox] = val;
        float sq = val * val;
#pragma unroll
        for (int off = 1; off < 16; off <<= 1)
            sq += __shfl_xor(sq, off, 32);   // stays within each 16-lane half
        if ((lane & 15) == 0)
            atomicAdd(&sumsq[b * COUT + cout], sq);
    }
}

// ---------------------------------------------------------------------------
// Fused: demod apply + 2x bilinear upsample + BN (B,H,W) statistics.
// Pow2 geometry -> shift/mask addressing. 2 global atomics per block.
// ---------------------------------------------------------------------------
__global__ void upsample_stats_kernel(const float* __restrict__ convBuf,
                                      const float* __restrict__ sumsq,
                                      float* __restrict__ outFeat,
                                      float* __restrict__ bnSum,
                                      float* __restrict__ bnSsq,
                                      int C, int H, int W, int w2sh) {
    const int H2 = 2 * H, W2 = 2 * W;
    const int b = blockIdx.z, c = blockIdx.y;
    const int i = blockIdx.x * blockDim.x + threadIdx.x;
    float val = 0.0f, sq = 0.0f;
    if (i < H2 * W2) {
        int oy = i >> w2sh, ox = i & (W2 - 1);
        float fy = 0.5f * oy - 0.25f;
        float fx = 0.5f * ox - 0.25f;
        int yf = (int)floorf(fy), xf = (int)floorf(fx);
        float wy = fy - yf, wx = fx - xf;
        int y0c = min(max(yf, 0), H - 1), y1c = min(max(yf + 1, 0), H - 1);
        int x0c = min(max(xf, 0), W - 1), x1c = min(max(xf + 1, 0), W - 1);
        const float* p = convBuf + ((size_t)b * C + c) * H * W;
        float v00 = p[y0c * W + x0c], v01 = p[y0c * W + x1c];
        float v10 = p[y1c * W + x0c], v11 = p[y1c * W + x1c];
        float bi = (1.0f - wy) * ((1.0f - wx) * v00 + wx * v01)
                 +          wy * ((1.0f - wx) * v10 + wx * v11);
        float d  = rsqrtf(sumsq[b * C + c] + 1e-8f);
        val = bi * d;
        outFeat[(((size_t)b * C + c) * H2 + oy) * W2 + ox] = val;
        sq = val * val;
    }
    __shared__ float sh[2];
    if (threadIdx.x == 0) { sh[0] = 0.0f; sh[1] = 0.0f; }
    __syncthreads();
    float s1 = val, s2 = sq;
    for (int off = 1; off < 32; off <<= 1) {
        s1 += __shfl_xor(s1, off, 32);
        s2 += __shfl_xor(s2, off, 32);
    }
    if ((threadIdx.x & 31) == 0) { atomicAdd(&sh[0], s1); atomicAdd(&sh[1], s2); }
    __syncthreads();
    if (threadIdx.x == 0) { atomicAdd(&bnSum[c], sh[0]); atomicAdd(&bnSsq[c], sh[1]); }
}

__global__ void bn_finalize_kernel(const float* __restrict__ bnSum,
                                   const float* __restrict__ bnSsq,
                                   const float* __restrict__ g,
                                   const float* __restrict__ bb,
                                   float* __restrict__ scale,
                                   float* __restrict__ shift,
                                   int C, float invN) {
    int c = blockIdx.x * blockDim.x + threadIdx.x;
    if (c >= C) return;
    float mean = bnSum[c] * invN;
    float var  = bnSsq[c] * invN - mean * mean;
    float sc   = g[c] * rsqrtf(var + 1e-5f);
    scale[c] = sc;
    shift[c] = bb[c] - mean * sc;
}

__global__ void bn_apply_relu_kernel(float* __restrict__ feat,
                                     const float* __restrict__ scale,
                                     const float* __restrict__ shift,
                                     int hwsh, int cmask, long long N) {
    long long i = (long long)blockIdx.x * blockDim.x + threadIdx.x;
    if (i >= N) return;
    int c = (int)(i >> hwsh) & cmask;
    float v = feat[i] * scale[c] + shift[c];
    feat[i] = v > 0.0f ? v : 0.0f;
}

// ---------------------------------------------------------------------------
template<int CIN, int COUT, int H, int W, int MT>
static void launch_conv(const float* act, int actBS, const float* s,
                        const _Float16* w, const float* cb,
                        float* convBuf, float* ssq, hipStream_t st) {
    constexpr int SW   = (W < 16) ? W : 16;
    constexpr int R    = 16 / SW;
    constexpr int BTOT = CIN * (R + 2) * (SW + 2);
    constexpr int BPAD = (BTOT + 31) & ~31;
    constexpr size_t smemB = (size_t)(BPAD + MT * 1024) * sizeof(_Float16);
    dim3 g(BATCH * H * W / 16, COUT / (16 * MT));
    conv_wmma_kernel<CIN, COUT, H, W, MT>
        <<<g, 32 * MT, smemB, st>>>(act, actBS, s, w, cb, convBuf, ssq);
}

extern "C" void kernel_launch(void* const* d_in, const int* in_sizes, int n_in,
                              void* d_out, int out_size, void* d_ws, size_t ws_size,
                              hipStream_t stream) {
    (void)in_sizes; (void)n_in; (void)out_size; (void)ws_size;

    const float* x   = (const float*)d_in[0];
    const float* cst = (const float*)d_in[1];
    const float* fcw = (const float*)d_in[2];
    const float* fcb = (const float*)d_in[3];
    const float *modw[NLAYERS], *modb[NLAYERS], *convw[NLAYERS],
                *convb[NLAYERS], *bng[NLAYERS], *bnb[NLAYERS];
    for (int l = 0; l < NLAYERS; ++l) {
        modw[l]  = (const float*)d_in[4 + l];
        modb[l]  = (const float*)d_in[10 + l];
        convw[l] = (const float*)d_in[16 + l];
        convb[l] = (const float*)d_in[22 + l];
        bng[l]   = (const float*)d_in[28 + l];
        bnb[l]   = (const float*)d_in[34 + l];
    }
    float* out = (float*)d_out;
    char*  ws  = (char*)d_ws;

    static const int CIN[NLAYERS]  = {512, 256, 128, 64, 32, 16};
    static const int COUT[NLAYERS] = {256, 128, 64, 32, 16, 16};
    static const int HH[NLAYERS]   = {4, 8, 16, 32, 64, 128};

    size_t o = 0;
    auto alloc = [&](size_t bytes) -> size_t {
        size_t r = o;
        o = (o + bytes + 255) & ~(size_t)255;
        return r;
    };
    size_t styleOff = alloc((size_t)BATCH * 512 * 4);
    size_t sOff[NLAYERS], wOff[NLAYERS];
    int KR[NLAYERS], KP[NLAYERS];
    for (int l = 0; l < NLAYERS; ++l) {
        sOff[l] = alloc((size_t)BATCH * CIN[l] * 4);
        KR[l] = CIN[l] * 9;
        KP[l] = ((KR[l] + 31) / 32) * 32;
        wOff[l] = alloc((size_t)COUT[l] * KP[l] * 2);
    }
    size_t convOff  = alloc((size_t)BATCH * 16 * 128 * 128 * 4);
    size_t ssqOff   = alloc((size_t)BATCH * 256 * 4);
    size_t bnSumOff = alloc(256 * 4);
    size_t bnSsqOff = alloc(256 * 4);
    size_t bnScOff  = alloc(256 * 4);
    size_t bnShOff  = alloc(256 * 4);

    float* styleP = (float*)(ws + styleOff);
    float* convP  = (float*)(ws + convOff);
    float* ssqP   = (float*)(ws + ssqOff);
    float* bnSumP = (float*)(ws + bnSumOff);
    float* bnSsqP = (float*)(ws + bnSsqOff);
    float* bnScP  = (float*)(ws + bnScOff);
    float* bnShP  = (float*)(ws + bnShOff);

    style_kernel<<<(BATCH * 512 + 255) / 256, 256, 0, stream>>>(x, fcw, fcb, styleP);
    for (int l = 0; l < NLAYERS; ++l) {
        int n = BATCH * CIN[l];
        mod_kernel<<<(n + 255) / 256, 256, 0, stream>>>(
            styleP, modw[l], modb[l], (float*)(ws + sOff[l]), CIN[l]);
        int nw = COUT[l] * KP[l];
        pack_w_kernel<<<(nw + 255) / 256, 256, 0, stream>>>(
            convw[l], (_Float16*)(ws + wOff[l]), COUT[l], KR[l], KP[l]);
    }

    size_t featOff = 0;
    const float* act = cst;
    int actBS = 0;                       // batch stride 0 => broadcast const
    for (int l = 0; l < NLAYERS; ++l) {
        const int Cout = COUT[l];
        const int H = HH[l], W = HH[l];

        zero_kernel<<<(BATCH * Cout + 255) / 256, 256, 0, stream>>>(ssqP, BATCH * Cout);
        zero_kernel<<<1, 256, 0, stream>>>(bnSumP, Cout);
        zero_kernel<<<1, 256, 0, stream>>>(bnSsqP, Cout);

        const float* sP = (const float*)(ws + sOff[l]);
        const _Float16* wP = (const _Float16*)(ws + wOff[l]);
        switch (l) {
        case 0: launch_conv<512, 256,   4,   4, 4>(act, actBS, sP, wP, convb[l], convP, ssqP, stream); break;
        case 1: launch_conv<256, 128,   8,   8, 4>(act, actBS, sP, wP, convb[l], convP, ssqP, stream); break;
        case 2: launch_conv<128,  64,  16,  16, 4>(act, actBS, sP, wP, convb[l], convP, ssqP, stream); break;
        case 3: launch_conv< 64,  32,  32,  32, 2>(act, actBS, sP, wP, convb[l], convP, ssqP, stream); break;
        case 4: launch_conv< 32,  16,  64,  64, 1>(act, actBS, sP, wP, convb[l], convP, ssqP, stream); break;
        case 5: launch_conv< 16,  16, 128, 128, 1>(act, actBS, sP, wP, convb[l], convP, ssqP, stream); break;
        }

        const int H2 = 2 * H, W2 = 2 * W;
        dim3 gu((H2 * W2 + 255) / 256, Cout, BATCH);
        upsample_stats_kernel<<<gu, 256, 0, stream>>>(
            convP, ssqP, out + featOff, bnSumP, bnSsqP, Cout, H, W,
            __builtin_ctz((unsigned)W2));

        bn_finalize_kernel<<<(Cout + 63) / 64, 64, 0, stream>>>(
            bnSumP, bnSsqP, bng[l], bnb[l], bnScP, bnShP, Cout,
            1.0f / ((float)BATCH * H2 * W2));

        long long N = (long long)BATCH * Cout * H2 * W2;
        bn_apply_relu_kernel<<<(unsigned)((N + 255) / 256), 256, 0, stream>>>(
            out + featOff, bnScP, bnShP,
            __builtin_ctz((unsigned)(H2 * W2)), Cout - 1, N);

        act = out + featOff;
        actBS = Cout * H2 * W2;
        featOff += (size_t)N;
    }
}
